// Autoencoder_35845797052744
// MI455X (gfx1250) — matrix operand order repaired and checked by reading, not compiled, not run on hardware
//
#include <hip/hip_runtime.h>
#include <hip/hip_bf16.h>
#include <math.h>

// ---------------------------------------------------------------------------
// Types for CDNA5 WMMA / TDM (gfx1250, wave32)
// ---------------------------------------------------------------------------
typedef __attribute__((ext_vector_type(16))) __bf16        v16bf;
typedef __attribute__((ext_vector_type(8)))  float         v8f;
typedef __attribute__((ext_vector_type(4)))  unsigned int  u32x4;
typedef __attribute__((ext_vector_type(8)))  int           i32x8;
typedef __attribute__((ext_vector_type(4)))  int           i32x4;

__device__ __forceinline__ unsigned short f2bf(float f) {
    unsigned int u = __float_as_uint(f);
    unsigned int r = u + 0x7FFFu + ((u >> 16) & 1u);   // round-to-nearest-even
    return (unsigned short)(r >> 16);
}

__device__ __forceinline__ float gelu_exact(float x) {
    return 0.5f * x * (1.0f + erff(x * 0.70710678118654752f));
}

// ---------------------------------------------------------------------------
// Input packing: NCHW f32 -> NHWC bf16 (C padded to 4)
// ---------------------------------------------------------------------------
__global__ void pack_x_kernel(const float* __restrict__ x, unsigned short* __restrict__ out) {
    int i = blockIdx.x * blockDim.x + threadIdx.x;   // i over 16*256*256 pixels
    if (i >= 16 * 65536) return;
    int n = i >> 16, pix = i & 65535;
    unsigned short* o = out + (size_t)i * 4;
    #pragma unroll
    for (int c = 0; c < 3; ++c)
        o[c] = f2bf(x[((size_t)(n * 3 + c)) * 65536 + pix]);
    o[3] = 0;
}

// ---------------------------------------------------------------------------
// Weight packing -> bf16 B matrix [Kpad][Np], k = (ky*3+kx)*Cin + ci
// mode 0: enc conv    w[co][ci][ky][kx]
// mode 1: dec convT   B[k][co] = w[ci][co][2-ky][2-kx]  (flip+transpose folded in)
// ---------------------------------------------------------------------------
__global__ void pack_w_kernel(const float* __restrict__ w, unsigned short* __restrict__ Bm,
                              int mode, int Cin, int Cout, int Np, int Kpad) {
    int i = blockIdx.x * blockDim.x + threadIdx.x;
    if (i >= Kpad * Np) return;
    int n = i % Np, k = i / Np;
    float val = 0.0f;
    if (k < 9 * Cin && n < Cout) {
        int ci = k % Cin, tap = k / Cin;
        int ky = tap / 3, kx = tap - ky * 3;
        if (mode == 0) val = w[(((size_t)n * Cin + ci) * 3 + ky) * 3 + kx];
        else           val = w[(((size_t)ci * Cout + n) * 3 + (2 - ky)) * 3 + (2 - kx)];
    }
    Bm[i] = f2bf(val);
}

// emb (2048,512) f32 -> embT bf16 [512][2048]
__global__ void pack_embT_kernel(const float* __restrict__ emb, unsigned short* __restrict__ embT) {
    int i = blockIdx.x * blockDim.x + threadIdx.x;   // i = k*2048 + n
    if (i >= 512 * 2048) return;
    int n = i & 2047, k = i >> 11;
    embT[i] = f2bf(emb[(size_t)n * 512 + k]);
}

// ---------------------------------------------------------------------------
// Implicit-GEMM conv kernel using v_wmma_f32_16x16x32_bf16.
//   D[M][NP] = A(im2col of NHWC bf16 act) x B(packed weights)
// mode 0: stride-2 conv pad 1; mode 1: transposed conv (zero-dilated gather);
// mode 2: dense GEMM (VQ distances).
// B tile is streamed into LDS by the Tensor Data Mover (TENSOR_LOAD_TO_LDS,
// tracked with TENSORcnt); all 8 waves meanwhile build the im2col A tile.
// ---------------------------------------------------------------------------
struct GemmParams {
    const unsigned short* A;    // NHWC bf16 activations
    const unsigned short* Bm;   // [Kpad][Np] bf16
    float* Y;                   // [M][Np] f32 out (pre-BN)
    int mode;
    int cinShift;               // log2(Cin), or -1 => Cin==3
    int Cstride;                // channel stride of A (elements)
    int Hin, Win;
    int woutShift, houtShift;   // log2(Wout), log2(Hout)
    int Ktot, Kpad, M;
};

template <int WM, int WN, int MF, int NF, int NP>
__global__ __launch_bounds__(256)
void conv_gemm_kernel(GemmParams p) {
    constexpr int BM = WM * MF * 16;
    constexpr int BN = WN * NF * 16;
    __shared__ unsigned short ldsA[BM * 32] __attribute__((aligned(32)));  // A-frag layout
    __shared__ unsigned short ldsB[32 * BN] __attribute__((aligned(32)));  // row-major [k][n]

    const int tid   = threadIdx.x;
    const int lane  = tid & 31;
    const int wave  = tid >> 5;
    const int waveM = wave % WM;
    const int waveN = wave / WM;
    const int mtile = blockIdx.y * BM;
    const int ntile = blockIdx.x * BN;

    const int woutMask = (1 << p.woutShift) - 1;
    const int houtMask = (1 << p.houtShift) - 1;
    const int cinMask  = (p.cinShift >= 0) ? ((1 << p.cinShift) - 1) : 0;

    v8f acc[MF][NF];
    #pragma unroll
    for (int i = 0; i < MF; ++i)
        #pragma unroll
        for (int j = 0; j < NF; ++j)
            #pragma unroll
            for (int r = 0; r < 8; ++r) acc[i][j][r] = 0.0f;

    const unsigned ldsBoff = (unsigned)(size_t)(&ldsB[0]);   // LDS byte address for TDM D#
    const i32x4 z4 = {};
    const i32x8 z8 = {};

    for (int k0 = 0; k0 < p.Kpad; k0 += 32) {
        // ---- wave 0: kick off the Tensor Data Mover for the 32 x BN B tile ----
        if (wave == 0) {
            unsigned long long ga =
                (unsigned long long)(const char*)p.Bm + ((size_t)k0 * NP + ntile) * 2ull;
            unsigned rows = (unsigned)(p.Kpad - k0);
            u32x4 g0;
            g0.x = 1u;                                              // count=1, user desc
            g0.y = ldsBoff;                                         // lds_addr
            g0.z = (unsigned)(ga & 0xFFFFFFFFull);                  // global_addr[95:64]
            g0.w = (unsigned)((ga >> 32) & 0x01FFFFFFull) | (2u << 30);  // type=2 (image)
            i32x8 g1;
            g1[0] = (1 << 16);                                      // data_size=1 (2 bytes)
            g1[1] = (int)((NP & 0xFFFF) << 16);                     // tensor_dim0 lo16
            g1[2] = (int)(((NP >> 16) & 0xFFFF) | ((rows & 0xFFFF) << 16)); // dim0 hi | dim1 lo
            g1[3] = (int)(((rows >> 16) & 0xFFFF) | (BN << 16));    // dim1 hi | tile_dim0
            g1[4] = 32;                                             // tile_dim1=32, tile_dim2=0
            g1[5] = NP;                                             // tensor_dim0_stride lo32
            g1[6] = 0;
            g1[7] = 0;
            __builtin_amdgcn_tensor_load_to_lds(g0, g1, z4, z4, z8, 0);
        }

        // ---- all waves: stage A tile (BM x 32) in WMMA A-fragment order ----
        for (int e = tid; e < BM * 32; e += 256) {
            int ml = e >> 5;      // local row
            int kl = e & 31;      // local k
            int k  = k0 + kl;
            unsigned short hv = 0;
            if (k < p.Ktot) {
                int m = mtile + ml;
                if (p.mode == 2) {
                    hv = p.A[(size_t)m * p.Cstride + k];
                } else {
                    int ox = m & woutMask;  int t = m >> p.woutShift;
                    int oy = t & houtMask;  int n = t >> p.houtShift;
                    int ci, tap;
                    if (p.cinShift >= 0) { ci = k & cinMask; tap = k >> p.cinShift; }
                    else                 { tap = (int)((unsigned)k / 3u); ci = k - tap * 3; }
                    int ky = tap / 3, kx = tap - ky * 3;
                    int iy, ix; bool ok;
                    if (p.mode == 0) {                 // stride-2 conv, pad 1
                        iy = oy * 2 + ky - 1; ix = ox * 2 + kx - 1;
                        ok = (iy >= 0) & (iy < p.Hin) & (ix >= 0) & (ix < p.Win);
                    } else {                           // convT: zero-dilated gather
                        int ty = oy + ky - 1, tx = ox + kx - 1;
                        ok = (ty >= 0) & (tx >= 0) & !(ty & 1) & !(tx & 1);
                        iy = ty >> 1; ix = tx >> 1;
                        ok = ok & (iy < p.Hin) & (ix < p.Win);
                    }
                    if (ok) hv = p.A[(size_t)((n * p.Hin + iy) * p.Win + ix) * p.Cstride + ci];
                }
            }
            // A fragment layout: lane = (k%16)/8*16 + row%16 ; idx = (k/16)*8 + (k%8)
            int g = kl >> 4, k2 = kl & 15, half = k2 >> 3;
            int lidx = ((ml >> 4) * 512) + ((half * 16 + (ml & 15)) * 16) + (g * 8 + (k2 & 7));
            ldsA[lidx] = hv;
        }

        __builtin_amdgcn_s_wait_tensorcnt(0);   // TDM done (no-op for waves 1..7)
        __syncthreads();

        v16bf afr[MF], bfr[NF];
        #pragma unroll
        for (int mf = 0; mf < MF; ++mf)
            afr[mf] = *reinterpret_cast<const v16bf*>(&ldsA[(waveM * MF + mf) * 512 + lane * 16]);
        #pragma unroll
        for (int nf = 0; nf < NF; ++nf)
            #pragma unroll
            for (int s = 0; s < 16; ++s) {
                int kk = (s < 8) ? ((lane >> 4) * 8 + s) : (16 + (lane >> 4) * 8 + (s - 8));
                bfr[nf][s] = __builtin_bit_cast(__bf16, ldsB[kk * BN + (waveN * NF + nf) * 16 + (lane & 15)]);
            }

        #pragma unroll
        for (int mf = 0; mf < MF; ++mf)
            #pragma unroll
            for (int nf = 0; nf < NF; ++nf)
                acc[mf][nf] = __builtin_amdgcn_wmma_f32_16x16x32_bf16(
                    false, afr[mf], false, bfr[nf], (short)0, acc[mf][nf], false, false);

        __syncthreads();
    }

    // ---- store D: NP is compile-time => row strides fold into immediate offsets ----
    #pragma unroll
    for (int mf = 0; mf < MF; ++mf)
        #pragma unroll
        for (int nf = 0; nf < NF; ++nf) {
            int row0 = mtile + (waveM * MF + mf) * 16 + ((lane >> 4) << 3);
            int col  = ntile + (waveN * NF + nf) * 16 + (lane & 15);
            float* out = p.Y + (size_t)row0 * NP + col;
            #pragma unroll
            for (int r = 0; r < 8; ++r) out[(size_t)r * NP] = acc[mf][nf][r];
        }
}

// ---------------------------------------------------------------------------
// BatchNorm statistics (two-stage, deterministic)
// ---------------------------------------------------------------------------
__global__ void stats_partial_kernel(const float* __restrict__ Y, float* __restrict__ psum,
                                     float* __restrict__ psq, int M, int Np) {
    int ch = blockIdx.x * 64 + threadIdx.x;         // blockDim = 64, coalesced over channels
    int chunk = blockIdx.y;
    int m0 = chunk * 4096;
    int m1 = m0 + 4096; if (m1 > M) m1 = M;
    if (ch >= Np) return;
    float s = 0.0f, q = 0.0f;
    for (int m = m0; m < m1; ++m) {
        float v = Y[(size_t)m * Np + ch];
        s += v; q += v * v;
    }
    psum[chunk * 512 + ch] = s;
    psq [chunk * 512 + ch] = q;
}

__global__ void stats_final_kernel(const float* __restrict__ psum, const float* __restrict__ psq,
                                   const float* __restrict__ g, const float* __restrict__ be,
                                   float* __restrict__ scale, float* __restrict__ shift,
                                   int M, int Np, int Cout, int nchunk) {
    int ch = blockIdx.x * blockDim.x + threadIdx.x;
    if (ch >= Np) return;
    float s = 0.0f, q = 0.0f;
    for (int c = 0; c < nchunk; ++c) { s += psum[c * 512 + ch]; q += psq[c * 512 + ch]; }
    float inv  = 1.0f / (float)M;
    float mean = s * inv;
    float var  = fmaxf(q * inv - mean * mean, 0.0f);
    float rstd = rsqrtf(var + 1e-5f);
    float sc = 0.0f, sh = 0.0f;
    if (ch < Cout) { sc = g[ch] * rstd; sh = be[ch] - mean * sc; }  // padded channels -> exact 0
    scale[ch] = sc; shift[ch] = sh;
}

// BN + exact GELU, write bf16 NHWC activation (+ optional f32 copy for VQ)
__global__ void bn_act_kernel(const float* __restrict__ Y, const float* __restrict__ scale,
                              const float* __restrict__ shift, unsigned short* __restrict__ actout,
                              float* __restrict__ f32out, size_t total, int Np) {
    size_t i = (size_t)blockIdx.x * blockDim.x + threadIdx.x;
    if (i >= total) return;
    int ch = (int)(i % Np);
    float v = gelu_exact(Y[i] * scale[ch] + shift[ch]);
    if (actout) actout[i] = f2bf(v);
    if (f32out) f32out[i] = v;
}

// Final decoder epilogue: BN + GELU + tanh -> NCHW f32 d_out
__global__ void bn_tanh_out_kernel(const float* __restrict__ Y, const float* __restrict__ scale,
                                   const float* __restrict__ shift, float* __restrict__ out) {
    int i = blockIdx.x * blockDim.x + threadIdx.x;    // 16*3*256*256
    if (i >= 3145728) return;
    int n = i / 196608; int rem = i % 196608;
    int co = rem >> 16; int pix = rem & 65535;
    int m = (n << 16) + pix;
    float v = Y[(size_t)m * 16 + co] * scale[co] + shift[co];
    out[i] = tanhf(gelu_exact(v));
}

// ---------------------------------------------------------------------------
// VQ: codebook norms, argmin over d2 = ||e||^2 - 2*(f.e), loss
// ---------------------------------------------------------------------------
__global__ void emb_norm_kernel(const float* __restrict__ emb, float* __restrict__ enorm) {
    int n = blockIdx.x * blockDim.x + threadIdx.x;
    if (n >= 2048) return;
    float s = 0.0f;
    for (int k = 0; k < 512; ++k) { float v = emb[(size_t)n * 512 + k]; s += v * v; }
    enorm[n] = s;
}

__global__ void vq_argmin_kernel(const float* __restrict__ G, const float* __restrict__ enorm,
                                 int* __restrict__ idx) {
    __shared__ float sv[256];
    __shared__ int   si[256];
    int m = blockIdx.x, t = threadIdx.x;
    float best = 3.4e38f; int bi = 0x7FFFFFFF;
    for (int c = t; c < 2048; c += 256) {
        float d = enorm[c] - 2.0f * G[(size_t)m * 2048 + c];
        if (d < best) { best = d; bi = c; }
    }
    sv[t] = best; si[t] = bi; __syncthreads();
    for (int s = 128; s > 0; s >>= 1) {
        if (t < s) {
            if (sv[t + s] < sv[t] || (sv[t + s] == sv[t] && si[t + s] < si[t])) {
                sv[t] = sv[t + s]; si[t] = si[t + s];
            }
        }
        __syncthreads();
    }
    if (t == 0) idx[m] = si[0];
}

__global__ void vq_loss_partial_kernel(const float* __restrict__ h, const float* __restrict__ emb,
                                       const int* __restrict__ idx, float* __restrict__ part) {
    __shared__ float s[256];
    int m = blockIdx.x, t = threadIdx.x;
    const float* e = emb + (size_t)idx[m] * 512;
    float a = 0.0f;
    for (int c = t; c < 512; c += 256) { float d = h[(size_t)m * 512 + c] - e[c]; a += d * d; }
    s[t] = a; __syncthreads();
    for (int r = 128; r > 0; r >>= 1) { if (t < r) s[t] += s[t + r]; __syncthreads(); }
    if (t == 0) part[m] = s[0];
}

__global__ void vq_loss_final_kernel(const float* __restrict__ part, float* __restrict__ out) {
    __shared__ float s[256];
    int t = threadIdx.x;
    float a = 0.0f;
    for (int m = t; m < 4096; m += 256) a += part[m];
    s[t] = a; __syncthreads();
    for (int r = 128; r > 0; r >>= 1) { if (t < r) s[t] += s[t + r]; __syncthreads(); }
    if (t == 0) out[0] = 1.25f * s[0] / 2097152.0f;   // mean + 0.25*mean of (h-e)^2
}

// ---------------------------------------------------------------------------
// Host orchestration
// ---------------------------------------------------------------------------
static inline int ilog2i(int v) { int s = 0; while ((1 << s) < v) ++s; return s; }

extern "C" void kernel_launch(void* const* d_in, const int* in_sizes, int n_in,
                              void* d_out, int out_size, void* d_ws, size_t ws_size,
                              hipStream_t stream) {
    (void)in_sizes; (void)n_in; (void)out_size; (void)ws_size;

    const float* x   = (const float*)d_in[0];
    const float* emb = (const float*)d_in[33];

    // workspace carve-out
    char* ws = (char*)d_ws;
    size_t off = 0;
    auto carve = [&](size_t bytes) -> void* {
        void* p = ws + off;
        off = (off + bytes + 255) & ~(size_t)255;
        return p;
    };
    float*          Yb    = (float*)         carve(67108864);   // max 16.7M f32 (pre-BN / VQ dots)
    unsigned short* actA  = (unsigned short*)carve(33554432);   // ping
    unsigned short* actB  = (unsigned short*)carve(33554432);   // pong
    unsigned short* xpack = (unsigned short*)carve(8388608);
    float*          hf32  = (float*)         carve(8388608);    // encoder output (VQ loss)
    unsigned short* Bmat  = (unsigned short*)carve(4718592);
    unsigned short* embT  = (unsigned short*)carve(2097152);
    float*          psum  = (float*)         carve(524288);
    float*          psq   = (float*)         carve(524288);
    float*          scale = (float*)         carve(2048);
    float*          shift = (float*)         carve(2048);
    float*          enorm = (float*)         carve(8192);
    int*            vqidx = (int*)           carve(16384);
    float*          lpart = (float*)         carve(16384);

    auto launch_gemm = [&](const GemmParams& p, int Np, int M, bool smallN) {
        if (smallN) {
            conv_gemm_kernel<8, 1, 1, 1, 16><<<dim3(1, M / 128), 256, 0, stream>>>(p);
            return;
        }
        dim3 grid(Np / 64, M / 128);
        switch (Np) {
            case 64:   conv_gemm_kernel<4, 2, 2, 2, 64  ><<<grid, 256, 0, stream>>>(p); break;
            case 128:  conv_gemm_kernel<4, 2, 2, 2, 128 ><<<grid, 256, 0, stream>>>(p); break;
            case 256:  conv_gemm_kernel<4, 2, 2, 2, 256 ><<<grid, 256, 0, stream>>>(p); break;
            case 512:  conv_gemm_kernel<4, 2, 2, 2, 512 ><<<grid, 256, 0, stream>>>(p); break;
            case 2048: conv_gemm_kernel<4, 2, 2, 2, 2048><<<grid, 256, 0, stream>>>(p); break;
        }
    };

    auto run_layer = [&](int mode, const unsigned short* Ain, int Cin, int Cstride,
                         int Hin, int Win, int Hout, int Wout, int Cout, int Np,
                         const float* w, const float* g, const float* be,
                         unsigned short* actout, float* f32out, bool smallN) {
        int Ktot = 9 * Cin;
        int Kpad = (Ktot + 31) & ~31;
        int M = 16 * Hout * Wout;
        int totB = Kpad * Np;
        pack_w_kernel<<<(totB + 255) / 256, 256, 0, stream>>>(w, Bmat, mode, Cin, Cout, Np, Kpad);
        int cinShift = (Cin == 3) ? -1 : ilog2i(Cin);
        GemmParams p{Ain, Bmat, Yb, mode, cinShift, Cstride, Hin, Win,
                     ilog2i(Wout), ilog2i(Hout), Ktot, Kpad, M};
        launch_gemm(p, Np, M, smallN);
        int nchunk = (M + 4095) / 4096;
        stats_partial_kernel<<<dim3((Np + 63) / 64, nchunk), 64, 0, stream>>>(Yb, psum, psq, M, Np);
        stats_final_kernel<<<(Np + 255) / 256, 256, 0, stream>>>(psum, psq, g, be, scale, shift,
                                                                 M, Np, Cout, nchunk);
        if (actout || f32out) {
            size_t tot = (size_t)M * Np;
            bn_act_kernel<<<(unsigned)((tot + 255) / 256), 256, 0, stream>>>(
                Yb, scale, shift, actout, f32out, tot, Np);
        }
    };

    // ---- encoder ----
    pack_x_kernel<<<(16 * 65536 + 255) / 256, 256, 0, stream>>>(x, xpack);
    run_layer(0, xpack, 3, 4, 256, 256, 128, 128, 64, 64,
              (const float*)d_in[1],  (const float*)d_in[3],  (const float*)d_in[4],  actA, nullptr, false);
    run_layer(0, actA, 64, 64, 128, 128, 64, 64, 128, 128,
              (const float*)d_in[5],  (const float*)d_in[7],  (const float*)d_in[8],  actB, nullptr, false);
    run_layer(0, actB, 128, 128, 64, 64, 32, 32, 256, 256,
              (const float*)d_in[9],  (const float*)d_in[11], (const float*)d_in[12], actA, nullptr, false);
    run_layer(0, actA, 256, 256, 32, 32, 16, 16, 512, 512,
              (const float*)d_in[13], (const float*)d_in[15], (const float*)d_in[16], actB, hf32, false);

    // ---- vector quantizer: G = f . embT via WMMA, then argmin + loss ----
    pack_embT_kernel<<<(512 * 2048 + 255) / 256, 256, 0, stream>>>(emb, embT);
    {
        GemmParams pv{actB, embT, Yb, 2, 9, 512, 16, 16, 4, 4, 512, 512, 4096};
        launch_gemm(pv, 2048, 4096, false);
    }
    emb_norm_kernel<<<8, 256, 0, stream>>>(emb, enorm);
    vq_argmin_kernel<<<4096, 256, 0, stream>>>(Yb, enorm, vqidx);
    vq_loss_partial_kernel<<<4096, 256, 0, stream>>>(hf32, emb, vqidx, lpart);
    vq_loss_final_kernel<<<1, 256, 0, stream>>>(lpart, ((float*)d_out) + 3145728);

    // ---- decoder (input is encoder output h, per reference) ----
    run_layer(1, actB, 512, 512, 16, 16, 32, 32, 256, 256,
              (const float*)d_in[17], (const float*)d_in[19], (const float*)d_in[20], actA, nullptr, false);
    run_layer(1, actA, 256, 256, 32, 32, 64, 64, 128, 128,
              (const float*)d_in[21], (const float*)d_in[23], (const float*)d_in[24], actB, nullptr, false);
    run_layer(1, actB, 128, 128, 64, 64, 128, 128, 64, 64,
              (const float*)d_in[25], (const float*)d_in[27], (const float*)d_in[28], actA, nullptr, false);
    run_layer(1, actA, 64, 64, 128, 128, 256, 256, 3, 16,
              (const float*)d_in[29], (const float*)d_in[31], (const float*)d_in[32], nullptr, nullptr, true);
    bn_tanh_out_kernel<<<(3145728 + 255) / 256, 256, 0, stream>>>(Yb, scale, shift, (float*)d_out);
}
